// GNNModel_31224412242361
// MI455X (gfx1250) — matrix operand order repaired and checked by reading, not compiled, not run on hardware
//
#include <hip/hip_runtime.h>

// ---------------------------------------------------------------------------
// MI455X / gfx1250 fused GNN implementation (round 2).
// Per-edge MLP chains as bf16 WMMA (v_wmma_f32_16x16x32_bf16, fp32 accum) on
// 32-edge tiles staged in LDS; M=32 per wave -> two C accumulators share each
// B fragment; no persistent A registers (reloaded from LDS as b128 pairs) to
// keep VGPR pressure ~40 and avoid the spills seen in round 1.
// ---------------------------------------------------------------------------

typedef unsigned short u16t;
typedef __attribute__((ext_vector_type(16))) __bf16 v16bf;
typedef __attribute__((ext_vector_type(8)))  float  v8f;

union BFrag { v16bf v; u16t u[16]; uint4 q[2]; };

#define NN 50000
#define EE 640000LL
#define DEVI static __device__ __forceinline__

// ---- bf16 helpers (RNE) ----------------------------------------------------
DEVI u16t f2bf(float f) {
  unsigned u = __float_as_uint(f);
  u += 0x7FFFu + ((u >> 16) & 1u);
  return (u16t)(u >> 16);
}
DEVI unsigned pack2(float a, float b) {
  return (unsigned)f2bf(a) | ((unsigned)f2bf(b) << 16);
}
DEVI float bf2f(u16t h) { return __uint_as_float(((unsigned)h) << 16); }

DEVI void atomicMaxF(float* addr, float val) {
  unsigned* ua = (unsigned*)addr;
  unsigned old = *ua;
  while (__uint_as_float(old) < val) {
    unsigned assumed = old;
    old = atomicCAS(ua, assumed, __float_as_uint(val));
    if (old == assumed) break;
  }
}

// ---------------------------------------------------------------------------
// One dense layer on a 32-row tile (one wave32).
//   A tile: LDS bf16 row-major [32][DIN], row stride SA (padded, 16B-aligned)
//   W: packed bf16 B-fragments [NT][KB][32 lanes][16 contiguous values]
//   Out: LDS bf16 [32][DOUT] stride SO; optional fp32 global out (GOUT).
// ---------------------------------------------------------------------------
template<int DIN, int DOUT, int ACT, bool GOUT>
DEVI void mlp_layer32(const u16t* __restrict__ pw, const float* __restrict__ bias,
                      const u16t* __restrict__ At, int SA,
                      u16t* __restrict__ Ot, int SO,
                      int lane, float* __restrict__ gout) {
  constexpr int KB = DIN / 32;
  constexpr int NT = DOUT / 16;
  const int l = lane & 15, hi = lane >> 4;
  const u16t* arow0 = At + l * SA;          // rows 0..15  (both hi halves read row l)
  const u16t* arow1 = At + (16 + l) * SA;   // rows 16..31
#pragma unroll 1
  for (int nt = 0; nt < NT; ++nt) {
    v8f c0 = {0.f, 0.f, 0.f, 0.f, 0.f, 0.f, 0.f, 0.f};
    v8f c1 = {0.f, 0.f, 0.f, 0.f, 0.f, 0.f, 0.f, 0.f};
    const u16t* wp = pw + (long long)nt * KB * 512 + lane * 16;
#pragma unroll 2
    for (int kb = 0; kb < KB; ++kb) {
      BFrag b, a0, a1;
      const uint4* bp = (const uint4*)(wp + kb * 512);
      b.q[0] = bp[0];
      b.q[1] = bp[1];
      // A fragment = two contiguous 8-element (16 B) runs per lane.
      a0.q[0] = *(const uint4*)(arow0 + kb * 32 + hi * 16);
      a0.q[1] = *(const uint4*)(arow0 + kb * 32 + hi * 16 + 8);
      a1.q[0] = *(const uint4*)(arow1 + kb * 32 + hi * 16);
      a1.q[1] = *(const uint4*)(arow1 + kb * 32 + hi * 16 + 8);
      c0 = __builtin_amdgcn_wmma_f32_16x16x32_bf16(false, a0.v, false, b.v,
                                                   (short)0, c0, false, false);
      c1 = __builtin_amdgcn_wmma_f32_16x16x32_bf16(false, a1.v, false, b.v,
                                                   (short)0, c1, false, false);
    }
    const float bv = bias[nt * 16 + l];
#pragma unroll
    for (int r = 0; r < 8; ++r) {
      float x0 = c0[r] + bv;
      float x1 = c1[r] + bv;
      if (ACT == 1) { x0 = x0 > 0.f ? x0 : 0.f; x1 = x1 > 0.f ? x1 : 0.f; }
      const int M = r + hi * 8;               // C/D layout: row = vgpr + 8*hi
      Ot[M * SO + nt * 16 + l]        = f2bf(x0);
      Ot[(M + 16) * SO + nt * 16 + l] = f2bf(x1);
      if (GOUT) {
        gout[(long long)M * DOUT + nt * 16 + l]        = x0;
        gout[(long long)(M + 16) * DOUT + nt * 16 + l] = x1;
      }
    }
  }
}

// ---- pack fp32 [din][dout] row-major weight into WMMA B-fragment layout ----
__global__ void pack_w(const float* __restrict__ W, u16t* __restrict__ PW,
                       int din, int dout) {
  const int KB = din / 32;
  const long long total = (long long)din * dout;
  for (long long idx = (long long)blockIdx.x * blockDim.x + threadIdx.x;
       idx < total; idx += (long long)gridDim.x * blockDim.x) {
    const int j = (int)(idx & 15);
    const int lane = (int)((idx >> 4) & 31);
    const int f = (int)(idx >> 9);
    const int kb = f % KB, nt = f / KB;
    const int k = kb * 32 + (lane >> 4) * 16 + j;
    const int n = nt * 16 + (lane & 15);
    PW[idx] = f2bf(W[(long long)k * dout + n]);
  }
}

__global__ void fill_f32(float* __restrict__ p, float v, long long n) {
  long long i = (long long)blockIdx.x * blockDim.x + threadIdx.x;
  if (i < n) p[i] = v;
}

// ---------------------------------------------------------------------------
// Node preprocessing: x = LN(bn0, where(x_in[:,0]==-999, dummy, x_in))
// ---------------------------------------------------------------------------
__global__ void node_pre(const float* __restrict__ xin, const float* __restrict__ dmy,
                         const float* __restrict__ g, const float* __restrict__ b,
                         float* __restrict__ X) {
  const int n = (blockIdx.x * blockDim.x + threadIdx.x) >> 5;
  const int lane = threadIdx.x & 31;
  if (n >= NN) return;
  const float* row = xin + (long long)n * 64;
  const bool inval = (row[0] == -999.0f);
  float v0 = inval ? dmy[lane] : row[lane];
  float v1 = inval ? dmy[lane + 32] : row[lane + 32];
  float s = v0 + v1, q = v0 * v0 + v1 * v1;
#pragma unroll
  for (int o = 16; o > 0; o >>= 1) { s += __shfl_xor(s, o, 32); q += __shfl_xor(q, o, 32); }
  const float mu = s * (1.f / 64.f);
  const float rs = rsqrtf(q * (1.f / 64.f) - mu * mu + 1e-6f);
  X[(long long)n * 64 + lane]      = (v0 - mu) * rs * g[lane] + b[lane];
  X[(long long)n * 64 + lane + 32] = (v1 - mu) * rs * g[lane + 32] + b[lane + 32];
}

// LN + activation (MODE 0: leaky_relu(0.01), 1: relu), wave per node.
template<int D, int MODE>
__global__ void ln_act(const float* __restrict__ in, const float* __restrict__ g,
                       const float* __restrict__ b, float* __restrict__ out) {
  const int n = (blockIdx.x * blockDim.x + threadIdx.x) >> 5;
  const int lane = threadIdx.x & 31;
  if (n >= NN) return;
  constexpr int PT = D / 32;
  float v[PT];
  float s = 0.f, q = 0.f;
  const float* row = in + (long long)n * D;
#pragma unroll
  for (int k = 0; k < PT; ++k) { v[k] = row[lane + 32 * k]; s += v[k]; q += v[k] * v[k]; }
#pragma unroll
  for (int o = 16; o > 0; o >>= 1) { s += __shfl_xor(s, o, 32); q += __shfl_xor(q, o, 32); }
  const float mu = s / (float)D;
  const float rs = rsqrtf(q / (float)D - mu * mu + 1e-6f);
#pragma unroll
  for (int k = 0; k < PT; ++k) {
    const int j = lane + 32 * k;
    float y = (v[k] - mu) * rs * g[j] + b[j];
    y = (MODE == 0) ? (y > 0.f ? y : 0.01f * y) : (y > 0.f ? y : 0.f);
    out[(long long)n * D + j] = y;
  }
}

// ---------------------------------------------------------------------------
// Fused edge encoder + attention score (32 edges per wave):
//  LN(edge_attr) -> 128 relu -> 256 relu -> 128 (e_enc, fp32 out)
//  -> att1 64 relu -> att2 scalar score ; atomicMax per-dst for softmax.
// ---------------------------------------------------------------------------
__global__ void __launch_bounds__(64)
encoder_kernel(const float* __restrict__ eattr, const int* __restrict__ ei,
               const float* __restrict__ lng, const float* __restrict__ lnb,
               const u16t* __restrict__ pw1, const float* __restrict__ b1,
               const u16t* __restrict__ pw2, const float* __restrict__ b2,
               const u16t* __restrict__ pw3, const float* __restrict__ b3,
               const u16t* __restrict__ pwa, const float* __restrict__ ba,
               const float* __restrict__ att2w, const float* __restrict__ att2b,
               float* __restrict__ eenc, float* __restrict__ score,
               float* __restrict__ attmax) {
  extern __shared__ u16t lds[];
  constexpr int S0 = 264, S1 = 136;
  const int lane = threadIdx.x & 31;
  u16t* P0 = lds + (threadIdx.x >> 5) * 32 * (S0 + S1);
  u16t* P1 = P0 + 32 * S0;
  const int nw = (gridDim.x * blockDim.x) >> 5;
  const int ntiles = (int)(EE / 32);
  for (int t = (blockIdx.x * blockDim.x + threadIdx.x) >> 5; t < ntiles; t += nw) {
    const long long te = (long long)t * 32;
    // LayerNorm of edge_attr: one lane per edge row.
    const float* row = eattr + (te + lane) * 32;
    float v[32];
    float s = 0.f, q = 0.f;
#pragma unroll
    for (int j = 0; j < 32; ++j) { v[j] = row[j]; s += v[j]; q += v[j] * v[j]; }
    const float mu = s * (1.f / 32.f);
    const float rs = rsqrtf(q * (1.f / 32.f) - mu * mu + 1e-6f);
#pragma unroll
    for (int j = 0; j < 32; j += 2) {
      *(unsigned*)(P0 + lane * S0 + j) =
          pack2((v[j] - mu) * rs * lng[j] + lnb[j],
                (v[j + 1] - mu) * rs * lng[j + 1] + lnb[j + 1]);
    }
    mlp_layer32<32, 128, 1, false>(pw1, b1, P0, S0, P1, S1, lane, nullptr);
    mlp_layer32<128, 256, 1, false>(pw2, b2, P1, S1, P0, S0, lane, nullptr);
    mlp_layer32<256, 128, 0, true >(pw3, b3, P0, S0, P1, S1, lane, eenc + te * 128);
    mlp_layer32<128, 64, 1, false>(pwa, ba, P1, S1, P0, S0, lane, nullptr);
    // attention score: one lane per edge, 64-dot.
    float acc = att2b[0];
    const u16t* hr = P0 + lane * S0;
    for (int j = 0; j < 64; ++j) acc += bf2f(hr[j]) * att2w[j];
    const long long e = te + lane;
    score[e] = acc;
    atomicMaxF(attmax + ei[EE + e], acc);
  }
}

// pass 2 of every segment softmax: ex = exp(v - max[dst]); sum per dst.
__global__ void softmax_pass2(const float* __restrict__ v, const int* __restrict__ ei,
                              const float* __restrict__ smax, float* __restrict__ ex,
                              float* __restrict__ ssum) {
  long long e = (long long)blockIdx.x * blockDim.x + threadIdx.x;
  if (e < EE) {
    const int d = ei[EE + e];
    const float x = __expf(v[e] - smax[d]);
    ex[e] = x;
    atomicAdd(ssum + d, x);
  }
}

// ---------------------------------------------------------------------------
// Fused edge_conv pass A (32 edges per wave):
//  h = m3(relu(m2(relu(m1([x_dst,x_src,e_msg]))))) ; w = tanh(h @ s_w + b)
//  e_msg recomputed on the fly from e_enc * attention-alpha.
// ---------------------------------------------------------------------------
template<int H>
__global__ void __launch_bounds__(64)
conv_kernel(const float* __restrict__ x, const float* __restrict__ eenc,
            const int* __restrict__ ei,
            const float* __restrict__ exa, const float* __restrict__ asum,
            const u16t* __restrict__ pw1, const float* __restrict__ b1,
            const u16t* __restrict__ pw2, const float* __restrict__ b2,
            const u16t* __restrict__ pw3, const float* __restrict__ b3,
            const float* __restrict__ sw, const float* __restrict__ sb,
            u16t* __restrict__ hbuf, float* __restrict__ wbuf,
            float* __restrict__ cmax) {
  extern __shared__ u16t lds[];
  constexpr int S0 = 264, S1 = 136;
  const int lane = threadIdx.x & 31;
  u16t* P0 = lds + (threadIdx.x >> 5) * 32 * (S0 + S1);
  u16t* P1 = P0 + 32 * S0;
  const int nw = (gridDim.x * blockDim.x) >> 5;
  const int ntiles = (int)(EE / 32);
  for (int t = (blockIdx.x * blockDim.x + threadIdx.x) >> 5; t < ntiles; t += nw) {
    const long long te = (long long)t * 32;
    for (int r = 0; r < 32; ++r) {
      const long long e = te + r;
      const int d = ei[EE + e], si = ei[e];
      const float2 vd = ((const float2*)(x + (long long)d * 64))[lane];
      const float2 vs = ((const float2*)(x + (long long)si * 64))[lane];
      *(unsigned*)(P0 + r * S0 + 2 * lane)      = pack2(vd.x, vd.y);
      *(unsigned*)(P0 + r * S0 + 64 + 2 * lane) = pack2(vs.x, vs.y);
      const float al = exa[e] / (asum[d] + 1e-16f);
      const float2* er2 = (const float2*)(eenc + e * 128);
#pragma unroll
      for (int j2 = lane; j2 < 64; j2 += 32) {
        const float2 v2 = er2[j2];
        *(unsigned*)(P0 + r * S0 + 128 + 2 * j2) = pack2(v2.x * al, v2.y * al);
      }
    }
    mlp_layer32<256, H, 1, false>(pw1, b1, P0, S0, P1, S1, lane, nullptr);
    mlp_layer32<H,   H, 1, false>(pw2, b2, P1, S1, P0, S0, lane, nullptr);
    mlp_layer32<H,   H, 0, false>(pw3, b3, P0, S0, P1, S1, lane, nullptr);
    // h tile -> global (bf16, packed u32 stores)
    for (int r = 0; r < 32; ++r) {
      const long long e = te + r;
#pragma unroll
      for (int j = lane * 2; j < H; j += 64) {
        *(unsigned*)(hbuf + e * H + j) =
            (unsigned)P1[r * S1 + j] | ((unsigned)P1[r * S1 + j + 1] << 16);
      }
    }
    // gate scalar: one lane per edge.
    float acc = sb[0];
    const u16t* hr = P1 + lane * S1;
    for (int j = 0; j < H; ++j) acc += bf2f(hr[j]) * sw[j];
    const float wv = tanhf(acc);
    const long long e = te + lane;
    wbuf[e] = wv;
    atomicMaxF(cmax + ei[EE + e], wv);
  }
}

// pass C: x_out[dst] += h * ((1+w)/2) * alpha
template<int H>
__global__ void conv_scatter(const u16t* __restrict__ hbuf, const float* __restrict__ wbuf,
                             const float* __restrict__ cex, const float* __restrict__ csum,
                             const int* __restrict__ ei, float* __restrict__ xout) {
  constexpr int G = H / 4;
  long long tid = (long long)blockIdx.x * blockDim.x + threadIdx.x;
  long long e = tid / G;
  if (e >= EE) return;
  const int j = (int)(tid % G) * 4;
  const int d = ei[EE + e];
  const float coef = (1.f + wbuf[e]) * 0.5f * cex[e] / (csum[d] + 1e-16f);
  float* o = xout + (long long)d * H + j;
  const u16t* h = hbuf + e * H + j;
#pragma unroll
  for (int k = 0; k < 4; ++k) atomicAdd(o + k, bf2f(h[k]) * coef);
}

// degree count + per-dst sum of e_enc
__global__ void deg_efsum(const float* __restrict__ eenc, const int* __restrict__ ei,
                          float* __restrict__ efs, float* __restrict__ deg) {
  long long tid = (long long)blockIdx.x * blockDim.x + threadIdx.x;
  long long e = tid >> 5;
  if (e >= EE) return;
  const int g = (int)(tid & 31) * 4;
  const int d = ei[EE + e];
  if (g == 0) atomicAdd(deg + d, 1.f);
  float* o = efs + (long long)d * 128 + g;
  const float* s = eenc + e * 128 + g;
#pragma unroll
  for (int k = 0; k < 4; ++k) atomicAdd(o + k, s[k]);
}

// skip = x @ proj + b; gate = sigmoid(skip @ gw + gb); xf = g*skip + (1-g)*x2
__global__ void __launch_bounds__(128)
skip_gate_kernel(const float* __restrict__ X, const float* __restrict__ X2,
                 const float* __restrict__ psw, const float* __restrict__ psb,
                 const float* __restrict__ gw, const float* __restrict__ gb,
                 float* __restrict__ XF) {
  __shared__ float xr[64];
  __shared__ float sk[128];
  const int n = blockIdx.x, t = threadIdx.x;
  if (t < 64) xr[t] = X[(long long)n * 64 + t];
  __syncthreads();
  float a = psb[t];
  for (int k = 0; k < 64; ++k) a += xr[k] * psw[k * 128 + t];
  sk[t] = a;
  __syncthreads();
  float ga = gb[t];
  for (int k = 0; k < 128; ++k) ga += sk[k] * gw[k * 128 + t];
  const float gt = 1.f / (1.f + __expf(-ga));
  XF[(long long)n * 128 + t] = gt * sk[t] + (1.f - gt) * X2[(long long)n * 128 + t];
}

// node head: xc=[xf, ef_mean]; elu MLP 256->128->64->7; logits + softmax
__global__ void __launch_bounds__(128)
node_head_kernel(const float* __restrict__ XF, const float* __restrict__ efs,
                 const float* __restrict__ deg,
                 const float* __restrict__ w1, const float* __restrict__ b1,
                 const float* __restrict__ w2, const float* __restrict__ b2,
                 const float* __restrict__ w3, const float* __restrict__ b3,
                 float* __restrict__ out_logits, float* __restrict__ out_probs) {
  __shared__ float xc[256];
  __shared__ float h1[128];
  __shared__ float h2[64];
  __shared__ float lg[7];
  const int n = blockIdx.x, t = threadIdx.x;
  float dg = deg[n];
  dg = dg < 1.f ? 1.f : dg;
  xc[t]       = XF[(long long)n * 128 + t];
  xc[128 + t] = efs[(long long)n * 128 + t] / dg;
  __syncthreads();
  float a = b1[t];
  for (int k = 0; k < 256; ++k) a += xc[k] * w1[k * 128 + t];
  h1[t] = a > 0.f ? a : (__expf(a) - 1.f);
  __syncthreads();
  if (t < 64) {
    float a2 = b2[t];
    for (int k = 0; k < 128; ++k) a2 += h1[k] * w2[k * 64 + t];
    h2[t] = a2 > 0.f ? a2 : (__expf(a2) - 1.f);
  }
  __syncthreads();
  if (t < 7) {
    float a3 = b3[t];
    for (int k = 0; k < 64; ++k) a3 += h2[k] * w3[k * 7 + t];
    lg[t] = a3;
    out_logits[(long long)n * 7 + t] = a3;
  }
  __syncthreads();
  if (t < 7) {
    float m = lg[0];
    for (int c = 1; c < 7; ++c) m = fmaxf(m, lg[c]);
    float ssum = 0.f;
    for (int c = 0; c < 7; ++c) ssum += __expf(lg[c] - m);
    out_probs[(long long)n * 7 + t] = __expf(lg[t] - m) / ssum;
  }
}

// ---------------------------------------------------------------------------
// Fused edge predictor: [xf_src, xf_dst, e_enc] -> 128 relu -> 64 relu -> 1
// 32 edges per wave; 1 wave per block (LDS 33.8 KB).
// ---------------------------------------------------------------------------
__global__ void __launch_bounds__(32)
edgepred_kernel(const float* __restrict__ xf, const float* __restrict__ eenc,
                const int* __restrict__ ei,
                const u16t* __restrict__ pw1, const float* __restrict__ b1,
                const u16t* __restrict__ pw2, const float* __restrict__ b2,
                const float* __restrict__ ep3w, const float* __restrict__ ep3b,
                float* __restrict__ out_logit, float* __restrict__ out_prob) {
  extern __shared__ u16t lds[];
  constexpr int S0 = 392, S1 = 136;
  const int lane = threadIdx.x & 31;
  u16t* P0 = lds;
  u16t* P1 = P0 + 32 * S0;
  const int nw = (gridDim.x * blockDim.x) >> 5;
  const int ntiles = (int)(EE / 32);
  for (int t = (blockIdx.x * blockDim.x + threadIdx.x) >> 5; t < ntiles; t += nw) {
    const long long te = (long long)t * 32;
    for (int r = 0; r < 32; ++r) {
      const long long e = te + r;
      const int si = ei[e], d = ei[EE + e];
      const float2* a0 = (const float2*)(xf + (long long)si * 128);
      const float2* a1 = (const float2*)(xf + (long long)d * 128);
      const float2* a2 = (const float2*)(eenc + e * 128);
#pragma unroll
      for (int j2 = lane; j2 < 64; j2 += 32) {
        const float2 v0 = a0[j2], v1 = a1[j2], v2 = a2[j2];
        *(unsigned*)(P0 + r * S0 + 2 * j2)       = pack2(v0.x, v0.y);
        *(unsigned*)(P0 + r * S0 + 128 + 2 * j2) = pack2(v1.x, v1.y);
        *(unsigned*)(P0 + r * S0 + 256 + 2 * j2) = pack2(v2.x, v2.y);
      }
    }
    mlp_layer32<384, 128, 1, false>(pw1, b1, P0, S0, P1, S1, lane, nullptr);
    mlp_layer32<128, 64, 1, false>(pw2, b2, P1, S1, P0, S0, lane, nullptr);
    float acc = ep3b[0];
    const u16t* hr = P0 + lane * S0;
    for (int j = 0; j < 64; ++j) acc += bf2f(hr[j]) * ep3w[j];
    const long long e = te + lane;
    out_logit[e] = acc;
    out_prob[e]  = 1.f / (1.f + __expf(-acc));
  }
}

// ---------------------------------------------------------------------------
extern "C" void kernel_launch(void* const* d_in, const int* in_sizes, int n_in,
                              void* d_out, int out_size, void* d_ws, size_t ws_size,
                              hipStream_t stream) {
  (void)in_sizes; (void)n_in; (void)out_size; (void)ws_size;
  // Inputs, recursive dict-insertion order of setup_inputs()/make_params().
  const float* x_in  = (const float*)d_in[0];
  const int*   ei    = (const int*)d_in[1];      // [2,E]: row0 src, row1 dst
  const float* eattr = (const float*)d_in[2];
  const float* dummy = (const float*)d_in[3];
  const float* bn0g  = (const float*)d_in[4];  const float* bn0b  = (const float*)d_in[5];
  const float* psw   = (const float*)d_in[6];  const float* psb   = (const float*)d_in[7];
  const float* elng  = (const float*)d_in[8];  const float* elnb  = (const float*)d_in[9];
  const float* enc1w = (const float*)d_in[10]; const float* enc1b = (const float*)d_in[11];
  const float* enc2w = (const float*)d_in[12]; const float* enc2b = (const float*)d_in[13];
  const float* enc3w = (const float*)d_in[14]; const float* enc3b = (const float*)d_in[15];
  const float* att1w = (const float*)d_in[16]; const float* att1b = (const float*)d_in[17];
  const float* att2w = (const float*)d_in[18]; const float* att2b = (const float*)d_in[19];
  const float* c1m1w = (const float*)d_in[20]; const float* c1m1b = (const float*)d_in[21];
  const float* c1m2w = (const float*)d_in[22]; const float* c1m2b = (const float*)d_in[23];
  const float* c1m3w = (const float*)d_in[24]; const float* c1m3b = (const float*)d_in[25];
  const float* c1sw  = (const float*)d_in[26]; const float* c1sb  = (const float*)d_in[27];
  const float* c2m1w = (const float*)d_in[28]; const float* c2m1b = (const float*)d_in[29];
  const float* c2m2w = (const float*)d_in[30]; const float* c2m2b = (const float*)d_in[31];
  const float* c2m3w = (const float*)d_in[32]; const float* c2m3b = (const float*)d_in[33];
  const float* c2sw  = (const float*)d_in[34]; const float* c2sb  = (const float*)d_in[35];
  const float* bn1g  = (const float*)d_in[36]; const float* bn1b  = (const float*)d_in[37];
  const float* bn2g  = (const float*)d_in[38]; const float* bn2b  = (const float*)d_in[39];
  const float* gw    = (const float*)d_in[40]; const float* gb    = (const float*)d_in[41];
  const float* np1w  = (const float*)d_in[42]; const float* np1b  = (const float*)d_in[43];
  const float* np2w  = (const float*)d_in[44]; const float* np2b  = (const float*)d_in[45];
  const float* np3w  = (const float*)d_in[46]; const float* np3b  = (const float*)d_in[47];
  const float* ep1w  = (const float*)d_in[48]; const float* ep1b  = (const float*)d_in[49];
  const float* ep2w  = (const float*)d_in[50]; const float* ep2b  = (const float*)d_in[51];
  const float* ep3w  = (const float*)d_in[52]; const float* ep3b  = (const float*)d_in[53];

  float* out_nlog = (float*)d_out;                          // [N,7]
  float* out_elog = out_nlog + (long long)NN * 7;           // [E]
  float* out_npr  = out_elog + EE;                          // [N,7]
  float* out_epr  = out_npr + (long long)NN * 7;            // [E]

  // deterministic bump allocator over d_ws
  char* wsb = (char*)d_ws;
  size_t cur = 0;
  auto alloc = [&](size_t bytes) -> void* {
    void* p = (void*)(wsb + cur);
    cur += (bytes + 255) & ~(size_t)255;
    return p;
  };
  float* X    = (float*)alloc((size_t)NN * 64 * 4);
  float* EENC = (float*)alloc((size_t)EE * 128 * 4);
  float* SC   = (float*)alloc((size_t)EE * 4);
  float* EXA  = (float*)alloc((size_t)EE * 4);
  float* AMAX = (float*)alloc((size_t)NN * 4);
  float* ASUM = (float*)alloc((size_t)NN * 4);
  float* CW   = (float*)alloc((size_t)EE * 4);
  float* CEX  = (float*)alloc((size_t)EE * 4);
  float* CMAX = (float*)alloc((size_t)NN * 4);
  float* CSUM = (float*)alloc((size_t)NN * 4);
  u16t*  HBUF = (u16t*)alloc((size_t)EE * 128 * 2);
  float* X1R  = (float*)alloc((size_t)NN * 64 * 4);
  float* X1   = (float*)alloc((size_t)NN * 64 * 4);
  float* X2R  = (float*)alloc((size_t)NN * 128 * 4);
  float* X2   = (float*)alloc((size_t)NN * 128 * 4);
  float* XF   = (float*)alloc((size_t)NN * 128 * 4);
  float* DEG  = (float*)alloc((size_t)NN * 4);
  float* EFS  = (float*)alloc((size_t)NN * 128 * 4);

  struct PackInfo { const float* w; int din, dout; u16t* pw; };
  PackInfo packs[12] = {
    {enc1w, 32, 128, nullptr}, {enc2w, 128, 256, nullptr}, {enc3w, 256, 128, nullptr},
    {att1w, 128, 64, nullptr},
    {c1m1w, 256, 64, nullptr}, {c1m2w, 64, 64, nullptr}, {c1m3w, 64, 64, nullptr},
    {c2m1w, 256, 128, nullptr}, {c2m2w, 128, 128, nullptr}, {c2m3w, 128, 128, nullptr},
    {ep1w, 384, 128, nullptr}, {ep2w, 128, 64, nullptr}
  };
  for (int i = 0; i < 12; ++i)
    packs[i].pw = (u16t*)alloc((size_t)packs[i].din * packs[i].dout * 2);
  for (int i = 0; i < 12; ++i) {
    long long total = (long long)packs[i].din * packs[i].dout;
    pack_w<<<(int)((total + 255) / 256), 256, 0, stream>>>(packs[i].w, packs[i].pw,
                                                           packs[i].din, packs[i].dout);
  }

  // 1) node preprocessing
  node_pre<<<(NN * 32 + 255) / 256, 256, 0, stream>>>(x_in, dummy, bn0g, bn0b, X);

  // 2) edge encoder + attention softmax
  fill_f32<<<(NN + 255) / 256, 256, 0, stream>>>(AMAX, -1e30f, NN);
  hipMemsetAsync(ASUM, 0, (size_t)NN * 4, stream);
  const size_t lds_std = 2 * 32 * (264 + 136) * sizeof(u16t);   // 51200 B
  encoder_kernel<<<2048, 64, lds_std, stream>>>(eattr, ei, elng, elnb,
      packs[0].pw, enc1b, packs[1].pw, enc2b, packs[2].pw, enc3b,
      packs[3].pw, att1b, att2w, att2b, EENC, SC, AMAX);
  softmax_pass2<<<(int)((EE + 255) / 256), 256, 0, stream>>>(SC, ei, AMAX, EXA, ASUM);

  // 3) edge_conv1 (H=64)
  hipMemsetAsync(X1R, 0, (size_t)NN * 64 * 4, stream);
  fill_f32<<<(NN + 255) / 256, 256, 0, stream>>>(CMAX, -1e30f, NN);
  hipMemsetAsync(CSUM, 0, (size_t)NN * 4, stream);
  conv_kernel<64><<<2048, 64, lds_std, stream>>>(X, EENC, ei, EXA, ASUM,
      packs[4].pw, c1m1b, packs[5].pw, c1m2b, packs[6].pw, c1m3b, c1sw, c1sb,
      HBUF, CW, CMAX);
  softmax_pass2<<<(int)((EE + 255) / 256), 256, 0, stream>>>(CW, ei, CMAX, CEX, CSUM);
  conv_scatter<64><<<(int)((EE * 16 + 255) / 256), 256, 0, stream>>>(HBUF, CW, CEX, CSUM, ei, X1R);
  ln_act<64, 0><<<(NN * 32 + 255) / 256, 256, 0, stream>>>(X1R, bn1g, bn1b, X1);

  // 4) edge_conv2 (H=128)
  hipMemsetAsync(X2R, 0, (size_t)NN * 128 * 4, stream);
  fill_f32<<<(NN + 255) / 256, 256, 0, stream>>>(CMAX, -1e30f, NN);
  hipMemsetAsync(CSUM, 0, (size_t)NN * 4, stream);
  conv_kernel<128><<<2048, 64, lds_std, stream>>>(X1, EENC, ei, EXA, ASUM,
      packs[7].pw, c2m1b, packs[8].pw, c2m2b, packs[9].pw, c2m3b, c2sw, c2sb,
      HBUF, CW, CMAX);
  softmax_pass2<<<(int)((EE + 255) / 256), 256, 0, stream>>>(CW, ei, CMAX, CEX, CSUM);
  conv_scatter<128><<<(int)((EE * 32 + 255) / 256), 256, 0, stream>>>(HBUF, CW, CEX, CSUM, ei, X2R);
  ln_act<128, 1><<<(NN * 32 + 255) / 256, 256, 0, stream>>>(X2R, bn2g, bn2b, X2);

  // 5) skip / gate fusion
  skip_gate_kernel<<<NN, 128, 0, stream>>>(X, X2, psw, psb, gw, gb, XF);

  // 6) per-dst mean of e_enc
  hipMemsetAsync(DEG, 0, (size_t)NN * 4, stream);
  hipMemsetAsync(EFS, 0, (size_t)NN * 128 * 4, stream);
  deg_efsum<<<(int)((EE * 32 + 255) / 256), 256, 0, stream>>>(EENC, ei, EFS, DEG);

  // 7) node head (logits + softmax)
  node_head_kernel<<<NN, 128, 0, stream>>>(XF, EFS, DEG, np1w, np1b, np2w, np2b,
                                           np3w, np3b, out_nlog, out_npr);

  // 8) edge predictor (logits + sigmoid)
  const size_t lds_pred = 32 * (392 + 136) * sizeof(u16t);      // 33792 B
  edgepred_kernel<<<2048, 32, lds_pred, stream>>>(XF, EENC, ei,
      packs[10].pw, ep1b, packs[11].pw, ep2b, ep3w, ep3b, out_elog, out_epr);
}
